// KimiMLAAttention_43963285242610
// MI455X (gfx1250) — matrix operand order repaired and checked by reading, not compiled, not run on hardware
//
#include <hip/hip_runtime.h>

#define L_SEQ 2048
#define HIDDEN 2048
#define NHEAD 16
#define QK_DIM 192
#define NOPE_D 128
#define ROPE_D 64
#define V_DIM 128
#define LORA_D 512
#define SCALE_F 0.07216878364870322f   // 192^-0.5
#define NEG_BIG (-3.0e38f)

typedef __attribute__((ext_vector_type(4)))  unsigned int u32x4;
typedef __attribute__((ext_vector_type(16))) __bf16      bf16x16;
typedef __attribute__((ext_vector_type(8)))  float       f32x8;

union Frag { u32x4 q[2]; bf16x16 v; };

__device__ __forceinline__ f32x8 zero8() {
  f32x8 z;
#pragma unroll
  for (int i = 0; i < 8; ++i) z[i] = 0.0f;
  return z;
}

__device__ __forceinline__ u32x4 zero4() {
  u32x4 z;
#pragma unroll
  for (int i = 0; i < 4; ++i) z[i] = 0u;
  return z;
}

__device__ __forceinline__ f32x8 wmma_bf16f32(const Frag& a, const Frag& b, f32x8 c) {
  return __builtin_amdgcn_wmma_f32_16x16x32_bf16(false, a.v, false, b.v, (short)0, c,
                                                 false, false);
}

// f32 -> bf16 bits, round-to-nearest-even
__device__ __forceinline__ unsigned short f2bfu(float f) {
  unsigned int u = __builtin_bit_cast(unsigned int, f);
  return (unsigned short)((u + 0x7fffu + ((u >> 16) & 1u)) >> 16);
}

// Async copy 16B global -> LDS (per-lane addresses), tracked by ASYNCcnt.
// Low 32 bits of a generic LDS pointer are the LDS byte address (ISA 10.2).
__device__ __forceinline__ void async_g2l_b128(void* lds_dst, const void* gsrc) {
  const unsigned dst = (unsigned)(unsigned long long)lds_dst;
  const unsigned long long src = (unsigned long long)gsrc;
  asm volatile("global_load_async_to_lds_b128 %0, %1, off" ::"v"(dst), "v"(src)
               : "memory");
}

__device__ __forceinline__ void wait_asynccnt0() {
  asm volatile("s_wait_asynccnt 0" ::: "memory");
}

// ---------------------------------------------------------------- convert
__global__ void cvt_kernel(const float* __restrict__ s, unsigned short* __restrict__ d,
                           long n) {
  long i = (long)blockIdx.x * 256 + threadIdx.x;
  const long stride = (long)gridDim.x * 256;
  for (; i < n; i += stride) d[i] = f2bfu(s[i]);
}

// W_unembed (H, V_DIM, LORA) f32 -> (H, LORA, V_DIM) bf16 (transpose per head)
__global__ void cvt_unembed_kernel(const float* __restrict__ s,
                                   unsigned short* __restrict__ d) {
  const long n = (long)NHEAD * LORA_D * V_DIM;
  long i = (long)blockIdx.x * 256 + threadIdx.x;
  const long stride = (long)gridDim.x * 256;
  for (; i < n; i += stride) {
    const long h  = i / ((long)LORA_D * V_DIM);
    const long r  = i % ((long)LORA_D * V_DIM);
    const long c  = r / V_DIM;
    const long vd = r % V_DIM;
    d[i] = f2bfu(s[(h * V_DIM + vd) * LORA_D + c]);
  }
}

// ---------------------------------------------------------------- GEMM (bf16 in, f32 acc)
// C = A(MxK,row) * B(KxN,row). 128x128 tile, 8 waves (2x4), BK=32,
// double-buffered LDS; A staged via GLOBAL_LOAD_ASYNC_TO_LDS_B128 (ASYNCcnt),
// B staged via VGPR with transposed store ([col][k], K-contiguous).
// OUT_MODE: 0 = f32 C[row*ldc+col], 1 = bf16 C[row*ldc+col], 2 = bf16 C[col*ldc+row]
template <int OUT_MODE>
__global__ __launch_bounds__(256) void gemm_kernel(
    const unsigned short* __restrict__ A, int lda, size_t sA,
    const unsigned short* __restrict__ B, int ldb, size_t sB,
    void* __restrict__ Cv, int ldc, size_t sC, int N, int K) {
  __shared__ unsigned short Alds[2 * 128 * 40];  // rows padded 32->40 (bank spread)
  __shared__ unsigned short Blds[2 * 128 * 40];  // [col][k]

  const int bz = blockIdx.z;
  A += (size_t)bz * sA;
  B += (size_t)bz * sB;
  const int row0 = blockIdx.y * 128;
  const int n0   = blockIdx.x * 128;
  const int t    = threadIdx.x;
  const int lane = t & 31, wave = t >> 5;
  const int wr = wave >> 2, wc = wave & 3;
  const int m = lane & 15, half = lane >> 4;

  f32x8 acc[4][2];
#pragma unroll
  for (int a = 0; a < 4; ++a)
#pragma unroll
    for (int b = 0; b < 2; ++b) acc[a][b] = zero8();

  auto stage = [&](int k0, int buf) {
    unsigned short* Ab = &Alds[buf * (128 * 40)];
    unsigned short* Bb = &Blds[buf * (128 * 40)];
#pragma unroll
    for (int it = 0; it < 2; ++it) {
      const int e = t * 8 + it * 2048;
      {  // A tile: 128x32 row-major, async 16B per lane
        const int r = e >> 5, c = e & 31;
        async_g2l_b128(&Ab[r * 40 + c], A + (size_t)(row0 + r) * lda + k0 + c);
      }
      {  // B tile: 32x128 global -> [col][k] in LDS (transpose, via VGPR)
        const int kr = e >> 7, c = e & 127;
        const int gcol = n0 + c;
        u32x4 v = zero4();
        if (gcol < N) v = *(const u32x4*)(B + (size_t)(k0 + kr) * ldb + gcol);
        const unsigned short* sp = (const unsigned short*)&v;
#pragma unroll
        for (int j = 0; j < 8; ++j) Bb[(c + j) * 40 + kr] = sp[j];
      }
    }
  };

  const int niter = K >> 5;
  int cur = 0;
  stage(0, 0);
  for (int i = 0; i < niter; ++i) {
    wait_asynccnt0();   // this wave's async A copies for buf[cur] done
    __syncthreads();    // + all waves' B ds_stores visible
    if (i + 1 < niter) stage((i + 1) << 5, cur ^ 1);

    const unsigned short* Ab = &Alds[cur * (128 * 40)];
    const unsigned short* Bb = &Blds[cur * (128 * 40)];
    Frag af[4], bf2[2];
#pragma unroll
    for (int tr = 0; tr < 4; ++tr) {  // A frag: row m, K {half*8..+7, 16+half*8..+7}
      const unsigned short* p = &Ab[(wr * 64 + tr * 16 + m) * 40 + half * 8];
      af[tr].q[0] = *(const u32x4*)p;
      af[tr].q[1] = *(const u32x4*)(p + 16);
    }
#pragma unroll
    for (int tc = 0; tc < 2; ++tc) {  // B frag: col m, K {half*8..+7, 16+half*8..+7}
      const unsigned short* p = &Bb[(wc * 32 + tc * 16 + m) * 40 + half * 8];
      bf2[tc].q[0] = *(const u32x4*)p;
      bf2[tc].q[1] = *(const u32x4*)(p + 16);
    }
#pragma unroll
    for (int tr = 0; tr < 4; ++tr)
#pragma unroll
      for (int tc = 0; tc < 2; ++tc)
        acc[tr][tc] = wmma_bf16f32(af[tr], bf2[tc], acc[tr][tc]);
    cur ^= 1;
  }

#pragma unroll
  for (int tr = 0; tr < 4; ++tr)
#pragma unroll
    for (int tc = 0; tc < 2; ++tc)
#pragma unroll
      for (int i = 0; i < 8; ++i) {
        const int row = row0 + wr * 64 + tr * 16 + i + 8 * half;
        const int col = n0 + wc * 32 + tc * 16 + m;
        if (col < N) {
          if (OUT_MODE == 0)
            ((float*)Cv)[(size_t)bz * sC + (size_t)row * ldc + col] = acc[tr][tc][i];
          else if (OUT_MODE == 1)
            ((unsigned short*)Cv)[(size_t)bz * sC + (size_t)row * ldc + col] =
                f2bfu(acc[tr][tc][i]);
          else
            ((unsigned short*)Cv)[(size_t)bz * sC + (size_t)col * ldc + row] =
                f2bfu(acc[tr][tc][i]);
        }
      }
}

// ---------------------------------------------------------------- RMSNorm + k_pe scatter
__global__ __launch_bounds__(256) void rmsnorm_kernel(const float* __restrict__ ckv,
                                                      const float* __restrict__ w,
                                                      unsigned short* __restrict__ latb,
                                                      unsigned short* __restrict__ kkb) {
  __shared__ float red[256];
  const int row = blockIdx.x;
  const int t = threadIdx.x;
  const float* src = ckv + (size_t)row * (LORA_D + ROPE_D);
  float s = 0.0f;
  for (int c = t; c < LORA_D; c += 256) {
    const float v = src[c];
    s += v * v;
  }
  red[t] = s;
  __syncthreads();
  for (int o = 128; o > 0; o >>= 1) {
    if (t < o) red[t] += red[t + o];
    __syncthreads();
  }
  const float rinv = rsqrtf(red[0] / (float)LORA_D + 1e-5f);
  for (int c = t; c < LORA_D; c += 256)
    latb[(size_t)row * LORA_D + c] = f2bfu(src[c] * rinv * w[c]);
  if (t < ROPE_D) {
    const unsigned short pe = f2bfu(src[LORA_D + t]);
#pragma unroll
    for (int h = 0; h < NHEAD; ++h)
      kkb[((size_t)h * L_SEQ + row) * QK_DIM + NOPE_D + t] = pe;
  }
}

// ---------------------------------------------------------------- flash attention
// grid (L/128, H), 256 thr = 8 waves; block-cooperative key loop:
// all waves walk the same 32-key chunks, K/V double-buffered in LDS via
// GLOBAL_LOAD_ASYNC_TO_LDS_B128; each wave owns 16 query rows.
// Fragment registers are software-pipelined (load d+1 while WMMA d).
__global__ __launch_bounds__(256) void attn_kernel(const unsigned short* __restrict__ qb,
                                                   const unsigned short* __restrict__ kk,
                                                   const unsigned short* __restrict__ vT,
                                                   unsigned short* __restrict__ aout) {
  __shared__ unsigned short Klds[2][32 * 200];   // [s_local][d], rows padded 192->200
  __shared__ unsigned short Vlds[2][128 * 40];   // [vd][s_local], rows padded 32->40
  __shared__ unsigned short Plds[8][16 * 40];

  const int h = blockIdx.y;
  const int t = threadIdx.x;
  const int wave = t >> 5;
  const int lane = t & 31;
  const int n = lane & 15;
  const int half = lane >> 4;
  const int l0 = blockIdx.x * 128 + wave * 16;

  // preload Q fragments (6 x 16x32 over 192 dims)
  Frag qf[6];
  const unsigned short* qrow = qb + (size_t)(l0 + n) * (NHEAD * QK_DIM) + h * QK_DIM;
#pragma unroll
  for (int d = 0; d < 6; ++d) {
    const unsigned short* p = qrow + d * 32 + half * 8;
    qf[d].q[0] = *(const u32x4*)p;
    qf[d].q[1] = *(const u32x4*)(p + 16);
  }

  f32x8 O[8];
#pragma unroll
  for (int v = 0; v < 8; ++v) O[v] = zero8();
  float mrow[8], lrow[8];
#pragma unroll
  for (int i = 0; i < 8; ++i) {
    mrow[i] = NEG_BIG;
    lrow[i] = 0.0f;
  }

  auto stageKV = [&](int s0, int buf) {
    // K chunk: 32 rows x 192 elems = 768 x 16B lane-copies over 3 passes
#pragma unroll
    for (int it = 0; it < 3; ++it) {
      const int j = t + it * 256;
      const int sl = j / 24;
      const int dp = (j % 24) * 8;
      async_g2l_b128(&Klds[buf][sl * 200 + dp],
                     kk + ((size_t)h * L_SEQ + (s0 + sl)) * QK_DIM + dp);
    }
    // V chunk: 128 rows x 32 elems = 512 x 16B lane-copies over 2 passes
#pragma unroll
    for (int it = 0; it < 2; ++it) {
      const int e = t * 8 + it * 2048;
      const int vd = e >> 5, sl = e & 31;
      async_g2l_b128(&Vlds[buf][vd * 40 + sl],
                     vT + ((size_t)h * V_DIM + vd) * L_SEQ + s0 + sl);
    }
    // depth-2 prefetch (speculative, global_prefetch_b8)
    if (s0 + 32 + 127 < blockIdx.x * 128 + 128) {
      __builtin_prefetch(kk + ((size_t)h * L_SEQ + (s0 + 32 + 2 * lane)) * QK_DIM, 0,
                         0);
    }
  };

  auto loadK = [&](const unsigned short* Kb, int d, Frag& b0, Frag& b1) {
    const unsigned short* kp0 = &Kb[n * 200 + d * 32 + half * 8];
    const unsigned short* kp1 = &Kb[(16 + n) * 200 + d * 32 + half * 8];
    b0.q[0] = *(const u32x4*)kp0;
    b0.q[1] = *(const u32x4*)(kp0 + 16);
    b1.q[0] = *(const u32x4*)kp1;
    b1.q[1] = *(const u32x4*)(kp1 + 16);
  };
  auto loadV = [&](const unsigned short* Vb, int v, Frag& vf) {
    const unsigned short* vp = &Vb[(v * 16 + n) * 40 + half * 8];
    vf.q[0] = *(const u32x4*)vp;
    vf.q[1] = *(const u32x4*)(vp + 16);
  };

  const int nch = ((blockIdx.x * 128 + 128) >> 5);  // uniform across block
  int cur = 0;
  stageKV(0, 0);
  for (int c = 0; c < nch; ++c) {
    const int s0 = c << 5;
    wait_asynccnt0();
    __syncthreads();
    if (c + 1 < nch) stageKV((c + 1) << 5, cur ^ 1);
    const unsigned short* Kb = Klds[cur];
    const unsigned short* Vb = Vlds[cur];

    // S = q · kk^T, K-fragments double-buffered in registers
    f32x8 S0 = zero8(), S1 = zero8();
    Frag kb[2][2];
    loadK(Kb, 0, kb[0][0], kb[0][1]);
#pragma unroll
    for (int d = 0; d < 6; ++d) {
      if (d + 1 < 6) loadK(Kb, d + 1, kb[(d + 1) & 1][0], kb[(d + 1) & 1][1]);
      S0 = wmma_bf16f32(qf[d], kb[d & 1][0], S0);
      S1 = wmma_bf16f32(qf[d], kb[d & 1][1], S1);
    }
    // online softmax update (rows live across 16-lane halves; xor<16 stays in half)
    float p0[8], p1[8], esc[8];
#pragma unroll
    for (int i = 0; i < 8; ++i) {
      const int lg = l0 + i + 8 * half;
      float v0 = S0[i] * SCALE_F;
      if (s0 + n > lg) v0 = NEG_BIG;
      float v1 = S1[i] * SCALE_F;
      if (s0 + 16 + n > lg) v1 = NEG_BIG;
      float mx = fmaxf(v0, v1);
#pragma unroll
      for (int off = 1; off < 16; off <<= 1) mx = fmaxf(mx, __shfl_xor(mx, off, 32));
      const float mnew = fmaxf(mrow[i], mx);
      const float e0 = __expf(v0 - mnew);
      const float e1 = __expf(v1 - mnew);
      float rs = e0 + e1;
#pragma unroll
      for (int off = 1; off < 16; off <<= 1) rs += __shfl_xor(rs, off, 32);
      esc[i] = __expf(mrow[i] - mnew);
      lrow[i] = lrow[i] * esc[i] + rs;
      mrow[i] = mnew;
      p0[i] = e0;
      p1[i] = e1;
    }
#pragma unroll
    for (int v = 0; v < 8; ++v)
#pragma unroll
      for (int i = 0; i < 8; ++i) O[v][i] *= esc[i];

    // P: C-layout -> A-layout via wave-private LDS tile
    unsigned short* pl = Plds[wave];
#pragma unroll
    for (int i = 0; i < 8; ++i) {
      pl[(i + 8 * half) * 40 + n] = f2bfu(p0[i]);
      pl[(i + 8 * half) * 40 + 16 + n] = f2bfu(p1[i]);
    }
    asm volatile("s_wait_dscnt 0" ::: "memory");
    Frag pf;
    {
      const unsigned short* p = pl + n * 40 + half * 8;
      pf.q[0] = *(const u32x4*)p;
      pf.q[1] = *(const u32x4*)(p + 16);
    }
    // O += P @ V, V-fragments double-buffered in registers
    Frag vb2[2];
    loadV(Vb, 0, vb2[0]);
#pragma unroll
    for (int v = 0; v < 8; ++v) {
      if (v + 1 < 8) loadV(Vb, v + 1, vb2[(v + 1) & 1]);
      O[v] = wmma_bf16f32(pf, vb2[v & 1], O[v]);
    }
    cur ^= 1;
  }

#pragma unroll
  for (int v = 0; v < 8; ++v)
#pragma unroll
    for (int i = 0; i < 8; ++i) {
      const int row = i + 8 * half;
      aout[(size_t)(l0 + row) * (NHEAD * V_DIM) + h * V_DIM + v * 16 + n] =
          f2bfu(O[v][i] / lrow[i]);
    }
}

// ---------------------------------------------------------------- launch
extern "C" void kernel_launch(void* const* d_in, const int* in_sizes, int n_in,
                              void* d_out, int out_size, void* d_ws, size_t ws_size,
                              hipStream_t stream) {
  (void)in_sizes; (void)n_in; (void)out_size; (void)ws_size;
  const float* x       = (const float*)d_in[0];
  // d_in[1] = mask (bool) — causal, handled analytically
  const float* Wq      = (const float*)d_in[2];
  const float* Wkv_a   = (const float*)d_in[3];
  const float* kv_ln_w = (const float*)d_in[4];
  const float* W_embed = (const float*)d_in[5];
  const float* W_unemb = (const float*)d_in[6];
  const float* Wo      = (const float*)d_in[7];

  char* base = (char*)d_ws;
  size_t off = 0;
  auto alloc = [&](size_t bytes) -> void* {
    void* p = base + off;
    off += bytes;
    off = (off + 255) & ~(size_t)255;
    return p;
  };

  unsigned short* xb    = (unsigned short*)alloc((size_t)L_SEQ * HIDDEN * 2);
  unsigned short* Wqb   = (unsigned short*)alloc((size_t)HIDDEN * NHEAD * QK_DIM * 2);
  unsigned short* qb2   = (unsigned short*)alloc((size_t)L_SEQ * NHEAD * QK_DIM * 2);
  unsigned short* Wkvab = (unsigned short*)alloc((size_t)HIDDEN * (LORA_D + ROPE_D) * 2);
  float*          ckvf  = (float*)alloc((size_t)L_SEQ * (LORA_D + ROPE_D) * 4);
  unsigned short* latb  = (unsigned short*)alloc((size_t)L_SEQ * LORA_D * 2);
  unsigned short* Web   = (unsigned short*)alloc((size_t)NHEAD * LORA_D * NOPE_D * 2);
  unsigned short* Wubt  = (unsigned short*)alloc((size_t)NHEAD * LORA_D * V_DIM * 2);
  unsigned short* kkb   = (unsigned short*)alloc((size_t)NHEAD * L_SEQ * QK_DIM * 2);
  unsigned short* vTb   = (unsigned short*)alloc((size_t)NHEAD * V_DIM * L_SEQ * 2);
  unsigned short* aoutb = (unsigned short*)alloc((size_t)L_SEQ * NHEAD * V_DIM * 2);
  unsigned short* Wob   = (unsigned short*)alloc((size_t)HIDDEN * HIDDEN * 2);

  dim3 blk(256);
  cvt_kernel<<<2048, blk, 0, stream>>>(x, xb, (long)L_SEQ * HIDDEN);
  cvt_kernel<<<2048, blk, 0, stream>>>(Wq, Wqb, (long)HIDDEN * NHEAD * QK_DIM);
  cvt_kernel<<<1024, blk, 0, stream>>>(Wkv_a, Wkvab, (long)HIDDEN * (LORA_D + ROPE_D));
  cvt_kernel<<<1024, blk, 0, stream>>>(W_embed, Web, (long)NHEAD * LORA_D * NOPE_D);
  cvt_kernel<<<2048, blk, 0, stream>>>(Wo, Wob, (long)HIDDEN * HIDDEN);
  cvt_unembed_kernel<<<1024, blk, 0, stream>>>(W_unemb, Wubt);

  // q = x @ Wq  -> bf16 (L, H*192)
  gemm_kernel<1><<<dim3(NHEAD * QK_DIM / 128, L_SEQ / 128, 1), blk, 0, stream>>>(
      xb, HIDDEN, 0, Wqb, NHEAD * QK_DIM, 0, qb2, NHEAD * QK_DIM, 0, NHEAD * QK_DIM,
      HIDDEN);
  // ckv = x @ Wkv_a -> f32 (L, 576)
  gemm_kernel<0><<<dim3((LORA_D + ROPE_D + 127) / 128, L_SEQ / 128, 1), blk, 0,
                   stream>>>(xb, HIDDEN, 0, Wkvab, LORA_D + ROPE_D, 0, ckvf,
                             LORA_D + ROPE_D, 0, LORA_D + ROPE_D, HIDDEN);
  // rmsnorm latent + scatter k_pe into kk[:, :, 128:192]
  rmsnorm_kernel<<<L_SEQ, blk, 0, stream>>>(ckvf, kv_ln_w, latb, kkb);
  // k_nope[h] = latent @ W_embed[h] -> kk[h][s][0:128]
  gemm_kernel<1><<<dim3(1, L_SEQ / 128, NHEAD), blk, 0, stream>>>(
      latb, LORA_D, 0, Web, NOPE_D, (size_t)LORA_D * NOPE_D, kkb, QK_DIM,
      (size_t)L_SEQ * QK_DIM, NOPE_D, LORA_D);
  // v[h] = latent @ W_unembed[h]^T, stored transposed -> vT[h][vd][s]
  gemm_kernel<2><<<dim3(1, L_SEQ / 128, NHEAD), blk, 0, stream>>>(
      latb, LORA_D, 0, Wubt, V_DIM, (size_t)LORA_D * V_DIM, vTb, L_SEQ,
      (size_t)V_DIM * L_SEQ, V_DIM, LORA_D);
  // flash attention per head -> aout (L, H*128) bf16
  attn_kernel<<<dim3(L_SEQ / 128, NHEAD), blk, 0, stream>>>(qb2, kkb, vTb, aoutb);
  // final: aout @ Wo -> f32 d_out
  gemm_kernel<0><<<dim3(HIDDEN / 128, L_SEQ / 128, 1), blk, 0, stream>>>(
      aoutb, HIDDEN, 0, Wob, HIDDEN, 0, (float*)d_out, HIDDEN, 0, HIDDEN, HIDDEN);
}
